// TacotronMelDecoder_395136991629
// MI455X (gfx1250) — compile-verified, matrix operand order checked by reading, not run on hardware
//
#include <hip/hip_runtime.h>
#include <hip/hip_bf16.h>

typedef __bf16 bf16_t;
typedef __attribute__((ext_vector_type(16))) __bf16 v16bf;
typedef __attribute__((ext_vector_type(8)))  float  v8f;

// ---------------------------------------------------------------------------
// WMMA helper: D = A(16x32 bf16) x B(32x16 bf16) + C(16x16 f32)
// ---------------------------------------------------------------------------
__device__ __forceinline__ v8f wmma_bf16(v16bf a, v16bf b, v8f c) {
  return __builtin_amdgcn_wmma_f32_16x16x32_bf16(
      /*neg_a=*/false, a, /*neg_b=*/false, b,
      /*c_mod=*/(short)0, c, /*reuse_a=*/false, /*reuse_b=*/false);
}

// Load a 16x32 bf16 fragment; source is K-contiguous row-major, leading dim ld
// (elements). Works for A ([M][K]) and B passed as Bt ([N][K]).
// ISA 16-bit A layout: lanes 0-15 hold K = k0+0..7 / k0+16..23, lanes 16-31
// hold K = k0+8..15 / k0+24..31  -> two contiguous 16B loads per lane.
__device__ __forceinline__ v16bf ld_frag(const bf16_t* __restrict__ base,
                                         int ld, int k0, int lane) {
  const int row = lane & 15;
  const int kb  = k0 + ((lane >> 4) << 3);
  const bf16_t* p = base + (size_t)row * ld + kb;
  union { v16bf v; uint4 q[2]; } f;
  f.q[0] = *(const uint4*)(p);
  f.q[1] = *(const uint4*)(p + 16);
  return f.v;
}

// ---------------------------------------------------------------------------
// Packing kernels: fp32 -> bf16, optional transpose + K zero-padding
// ---------------------------------------------------------------------------
__global__ void k_pack(const float* __restrict__ src, bf16_t* __restrict__ dst,
                       int R, int C, int ld) {
  int i = blockIdx.x * 256 + threadIdx.x;
  if (i >= R * ld) return;
  int r = i / ld, c = i % ld;
  dst[i] = (c < C) ? (bf16_t)src[(size_t)r * C + c] : (bf16_t)0.0f;
}

__global__ void k_packT(const float* __restrict__ src, bf16_t* __restrict__ dst,
                        int R, int C, int ld) {
  int i = blockIdx.x * 256 + threadIdx.x;
  if (i >= C * ld) return;
  int c = i / ld, r = i % ld;
  dst[i] = (r < R) ? (bf16_t)src[(size_t)r * C + c] : (bf16_t)0.0f;
}

// ---------------------------------------------------------------------------
// GEMM: C[M][N] f32 = A_bf16[M][ldA] @ Bt_bf16[N][ldB] + bias[N]
// block=(32,4); each wave owns a 16x64 output strip: one A fragment feeds
// four B tiles (5 fragment loads per 4 WMMA).
// ---------------------------------------------------------------------------
__global__ void __launch_bounds__(128) k_gemm_bias4(
    const bf16_t* __restrict__ A, const bf16_t* __restrict__ Bt,
    const float* __restrict__ bias, float* __restrict__ C,
    int N, int K, int ldA, int ldB) {
  const int lane = threadIdx.x;
  const int mt  = blockIdx.y * blockDim.y + threadIdx.y;
  const int nt0 = blockIdx.x * 4;
  const bf16_t* Abase = A + (size_t)mt * 16 * ldA;
  v8f acc[4] = {};
  for (int k0 = 0; k0 < K; k0 += 32) {
    v16bf a = ld_frag(Abase, ldA, k0, lane);
#pragma unroll
    for (int j = 0; j < 4; ++j)
      acc[j] = wmma_bf16(a, ld_frag(Bt + (size_t)(nt0 + j) * 16 * ldB, ldB, k0, lane),
                         acc[j]);
  }
  const int hi = lane >> 4;
#pragma unroll
  for (int j = 0; j < 4; ++j) {
    const int n = (nt0 + j) * 16 + (lane & 15);
    const float bv = bias[n];
#pragma unroll
    for (int r = 0; r < 8; ++r) {
      int m = mt * 16 + hi * 8 + r;
      C[(size_t)m * N + n] = acc[j][r] + bv;
    }
  }
}

// ---------------------------------------------------------------------------
// Persistent single-WGP LSTM recurrence (latency-critical path).
// h: LDS 16x512 bf16 (rows 8..15 zero). Wave w owns d in [16w,16w+16) and
// computes gate tiles i/f/g/o for that slice -> no cross-wave z exchange.
// Prefetches next step's xWk gate rows while WMMAs run.
// ---------------------------------------------------------------------------
__global__ void __launch_bounds__(1024, 1) k_lstm(
    const float* __restrict__ xWk,     // [B*Td][2048] = x@Wk + b (pre-biased)
    const bf16_t* __restrict__ WrT,    // [2048][512] (Wr transposed, bf16)
    bf16_t* __restrict__ hseq_bf,      // [B*Td][512]
    float* __restrict__ out_h,         // [8][512]
    float* __restrict__ out_c) {       // [8][512]
  __shared__ __align__(16) bf16_t h_bf[16][512];
  const int tid  = threadIdx.x;
  const int lane = tid & 31;
  const int wv   = tid >> 5;        // 0..31
  const int d0   = wv << 4;         // this wave's 16-wide d slice
  const int row  = lane & 15;
  const int hi   = lane >> 4;

  for (int i = tid; i < 16 * 512; i += 1024) (&h_bf[0][0])[i] = (bf16_t)0.0f;
  float c_reg[8];
#pragma unroll
  for (int r = 0; r < 8; ++r) c_reg[r] = 0.0f;
  __syncthreads();

  for (int t = 0; t < 256; ++t) {
    // warm next step's xWk gate rows (global_prefetch) while WMMAs execute
    if (lane < 8) {
      const int tn = (t + 1 < 256) ? t + 1 : t;
      const float* nx = xWk + (size_t)(lane * 256 + tn) * 2048 + d0;
      __builtin_prefetch(nx, 0, 0);
      __builtin_prefetch(nx + 512, 0, 0);
      __builtin_prefetch(nx + 1024, 0, 0);
      __builtin_prefetch(nx + 1536, 0, 0);
    }
    v8f acc_i = {}, acc_f = {}, acc_g = {}, acc_o = {};
    for (int k0 = 0; k0 < 512; k0 += 32) {
      const int kb = k0 + (hi << 3);
      union { v16bf v; uint4 q[2]; } a;            // A fragment from LDS
      a.q[0] = *(const uint4*)(&h_bf[row][kb]);
      a.q[1] = *(const uint4*)(&h_bf[row][kb + 16]);
      acc_i = wmma_bf16(a.v, ld_frag(WrT + (size_t)(0    + d0) * 512, 512, k0, lane), acc_i);
      acc_f = wmma_bf16(a.v, ld_frag(WrT + (size_t)(512  + d0) * 512, 512, k0, lane), acc_f);
      acc_g = wmma_bf16(a.v, ld_frag(WrT + (size_t)(1024 + d0) * 512, 512, k0, lane), acc_g);
      acc_o = wmma_bf16(a.v, ld_frag(WrT + (size_t)(1536 + d0) * 512, 512, k0, lane), acc_o);
    }
    __syncthreads();   // all waves done reading h_bf for this step
    if (hi == 0) {     // lanes 0..15 hold accumulator rows M = 0..7 (= batch)
      const int d = d0 + lane;
#pragma unroll
      for (int r = 0; r < 8; ++r) {
        const float* xr = xWk + (size_t)(r * 256 + t) * 2048;
        float zi = acc_i[r] + xr[d];
        float zf = acc_f[r] + xr[512 + d];
        float zg = acc_g[r] + xr[1024 + d];
        float zo = acc_o[r] + xr[1536 + d];
        float ig = 1.0f / (1.0f + __expf(-zi));
        float fg = 1.0f / (1.0f + __expf(-zf));
        float gg = tanhf(zg);
        float og = 1.0f / (1.0f + __expf(-zo));
        float c  = fg * c_reg[r] + ig * gg;
        float h  = og * tanhf(c);
        c_reg[r] = c;
        h_bf[r][d] = (bf16_t)h;
        hseq_bf[(size_t)(r * 256 + t) * 512 + d] = (bf16_t)h;
        if (t == 255) { out_h[r * 512 + d] = h; out_c[r * 512 + d] = c; }
      }
    }
    __syncthreads();   // h_bf updated before next step's reads
  }
}

// ---------------------------------------------------------------------------
// Additive-attention scores + softmax. One wave per (b,t) row.
// scores[e] = W3 . tanh(half1[b,e,:] + half2[b,t,:]) + b3, softmax over e.
// ---------------------------------------------------------------------------
__global__ void __launch_bounds__(256) k_scores(
    const float* __restrict__ half1,   // [B*Te][512]
    const float* __restrict__ half2,   // [B*Td][512]
    const float* __restrict__ W3, const float* __restrict__ b3,
    bf16_t* __restrict__ attn_bf) {    // [B*Td][256]
  const int lane = threadIdx.x & 31;
  const int wv   = threadIdx.x >> 5;         // 0..7
  const int rowv = blockIdx.x * 8 + wv;      // 0..2047 : b = rowv>>8
  const int b    = rowv >> 8;

  float h2[16], w3[16];
#pragma unroll
  for (int j = 0; j < 16; ++j) {
    h2[j] = half2[(size_t)rowv * 512 + lane + 32 * j];
    w3[j] = W3[lane + 32 * j];
  }
  __shared__ float sc[8][256];
  for (int e = 0; e < 256; ++e) {
    const float* h1 = half1 + (size_t)(b * 256 + e) * 512;
    float s = 0.0f;
#pragma unroll
    for (int j = 0; j < 16; ++j)
      s += w3[j] * tanhf(h1[lane + 32 * j] + h2[j]);
#pragma unroll
    for (int off = 16; off; off >>= 1) s += __shfl_xor(s, off, 32);
    if (lane == 0) sc[wv][e] = s + b3[0];
  }
  __syncthreads();
  float mx = -1e30f;
#pragma unroll
  for (int j = 0; j < 8; ++j) mx = fmaxf(mx, sc[wv][lane + 32 * j]);
#pragma unroll
  for (int off = 16; off; off >>= 1) mx = fmaxf(mx, __shfl_xor(mx, off, 32));
  float ex[8], sum = 0.0f;
#pragma unroll
  for (int j = 0; j < 8; ++j) { ex[j] = __expf(sc[wv][lane + 32 * j] - mx); sum += ex[j]; }
#pragma unroll
  for (int off = 16; off; off >>= 1) sum += __shfl_xor(sum, off, 32);
  const float inv = 1.0f / sum;
#pragma unroll
  for (int j = 0; j < 8; ++j)
    attn_bf[(size_t)rowv * 256 + lane + 32 * j] = (bf16_t)(ex[j] * inv);
}

// ---------------------------------------------------------------------------
// weighted[b] = attn[b] (256x256) @ attended[b] (256x512); 16x64 strip/wave.
// ---------------------------------------------------------------------------
__global__ void __launch_bounds__(128) k_weighted4(
    const bf16_t* __restrict__ attn, const bf16_t* __restrict__ attT,
    bf16_t* __restrict__ wgt_bf) {
  const int lane = threadIdx.x;
  const int bz  = blockIdx.z;
  const int mt  = blockIdx.y * blockDim.y + threadIdx.y;  // 0..15
  const int nt0 = blockIdx.x * 4;                         // 0,4,..,28
  const bf16_t* Abase = attn + (size_t)bz * 256 * 256 + (size_t)mt * 16 * 256;
  const bf16_t* Bb    = attT + (size_t)bz * 512 * 256;
  v8f acc[4] = {};
  for (int k0 = 0; k0 < 256; k0 += 32) {
    v16bf a = ld_frag(Abase, 256, k0, lane);
#pragma unroll
    for (int j = 0; j < 4; ++j)
      acc[j] = wmma_bf16(a, ld_frag(Bb + (size_t)(nt0 + j) * 16 * 256, 256, k0, lane),
                         acc[j]);
  }
  const int hi = lane >> 4;
#pragma unroll
  for (int j = 0; j < 4; ++j) {
    const int n = (nt0 + j) * 16 + (lane & 15);
#pragma unroll
    for (int r = 0; r < 8; ++r) {
      int m = bz * 256 + mt * 16 + hi * 8 + r;
      wgt_bf[(size_t)m * 512 + n] = (bf16_t)acc[j][r];
    }
  }
}

// ---------------------------------------------------------------------------
// out = concat([x, weighted]) @ Wout + bout as two K-halves; each wave owns a
// full 16x80 output row strip (all 5 N-tiles share one A fragment).
// ---------------------------------------------------------------------------
__global__ void __launch_bounds__(128) k_outproj(
    const bf16_t* __restrict__ Xbf, const bf16_t* __restrict__ Wbf,
    const bf16_t* __restrict__ WoutT, const float* __restrict__ bout,
    float* __restrict__ out) {
  const int lane = threadIdx.x;
  const int mt = blockIdx.y * blockDim.y + threadIdx.y;  // 0..127
  v8f acc[5] = {};
  for (int k0 = 0; k0 < 512; k0 += 32) {
    v16bf a = ld_frag(Xbf + (size_t)mt * 16 * 512, 512, k0, lane);
#pragma unroll
    for (int j = 0; j < 5; ++j)
      acc[j] = wmma_bf16(a, ld_frag(WoutT + (size_t)j * 16 * 1024, 1024, k0, lane),
                         acc[j]);
  }
  for (int k0 = 0; k0 < 512; k0 += 32) {
    v16bf a = ld_frag(Wbf + (size_t)mt * 16 * 512, 512, k0, lane);
#pragma unroll
    for (int j = 0; j < 5; ++j)
      acc[j] = wmma_bf16(a, ld_frag(WoutT + (size_t)j * 16 * 1024, 1024, 512 + k0, lane),
                         acc[j]);
  }
  const int hi = lane >> 4;
#pragma unroll
  for (int j = 0; j < 5; ++j) {
    const int n = j * 16 + (lane & 15);
    const float bv = bout[n];
#pragma unroll
    for (int r = 0; r < 8; ++r) {
      int m = mt * 16 + hi * 8 + r;
      out[(size_t)m * 80 + n] = acc[j][r] + bv;
    }
  }
}

// ---------------------------------------------------------------------------
extern "C" void kernel_launch(void* const* d_in, const int* in_sizes, int n_in,
                              void* d_out, int out_size, void* d_ws, size_t ws_size,
                              hipStream_t stream) {
  (void)in_sizes; (void)n_in; (void)out_size; (void)ws_size;
  const float* inputs   = (const float*)d_in[0];   // [8,256,80]
  const float* attended = (const float*)d_in[1];   // [8,256,512]
  const float* Wk       = (const float*)d_in[2];   // [80,2048]
  const float* Wr       = (const float*)d_in[3];   // [512,2048]
  const float* lb       = (const float*)d_in[4];   // [2048]
  const float* W1       = (const float*)d_in[5];   // [512,512]
  const float* b1       = (const float*)d_in[6];   // [512]
  const float* W3       = (const float*)d_in[7];   // [512,1]
  const float* b3       = (const float*)d_in[8];   // [1]
  const float* Wout     = (const float*)d_in[9];   // [1024,80]
  const float* bout     = (const float*)d_in[10];  // [80]
  float* out = (float*)d_out;                      // 163840 + 4096 + 4096

  char* ws = (char*)d_ws;
  size_t off = 0;
  auto alloc = [&](size_t bytes) -> void* {
    void* p = ws + off;
    off = (off + bytes + 255) & ~(size_t)255;
    return p;
  };
  float*  xWk     = (float*)  alloc(2048ull * 2048 * 4);
  float*  half1   = (float*)  alloc(2048ull * 512 * 4);
  float*  half2   = (float*)  alloc(2048ull * 512 * 4);
  bf16_t* hseq_bf = (bf16_t*) alloc(2048ull * 512 * 2);
  bf16_t* wgt_bf  = (bf16_t*) alloc(2048ull * 512 * 2);
  bf16_t* attn_bf = (bf16_t*) alloc(2048ull * 256 * 2);
  bf16_t* A_in    = (bf16_t*) alloc(2048ull * 96 * 2);
  bf16_t* WkT     = (bf16_t*) alloc(2048ull * 96 * 2);
  bf16_t* WrT     = (bf16_t*) alloc(2048ull * 512 * 2);
  bf16_t* W1T     = (bf16_t*) alloc(512ull * 512 * 2);
  bf16_t* WoutT   = (bf16_t*) alloc(80ull * 1024 * 2);
  bf16_t* att_bf  = (bf16_t*) alloc(2048ull * 512 * 2);
  bf16_t* attT    = (bf16_t*) alloc(8ull * 512 * 256 * 2);

  auto g1 = [](int n) { return (n + 255) / 256; };
  // fp32 -> bf16 packing / transposition (one-time, L2-resident afterwards)
  k_pack <<<g1(2048 * 96),  256, 0, stream>>>(inputs, A_in, 2048, 80, 96);
  k_packT<<<g1(2048 * 96),  256, 0, stream>>>(Wk, WkT, 80, 2048, 96);
  k_packT<<<g1(2048 * 512), 256, 0, stream>>>(Wr, WrT, 512, 2048, 512);
  k_packT<<<g1(512 * 512),  256, 0, stream>>>(W1, W1T, 512, 512, 512);
  k_packT<<<g1(80 * 1024),  256, 0, stream>>>(Wout, WoutT, 1024, 80, 1024);
  k_pack <<<g1(2048 * 512), 256, 0, stream>>>(attended, att_bf, 2048, 512, 512);
  for (int b = 0; b < 8; ++b)
    k_packT<<<g1(512 * 256), 256, 0, stream>>>(attended + (size_t)b * 256 * 512,
                                               attT + (size_t)b * 512 * 256,
                                               256, 512, 256);

  dim3 blk(32, 4);
  // xWk = inputs @ Wk + lstm_b  (K padded 80 -> 96)
  k_gemm_bias4<<<dim3(32, 32), blk, 0, stream>>>(A_in, WkT, lb, xWk, 2048, 96, 96, 96);
  // sequential LSTM recurrence (single persistent workgroup)
  k_lstm<<<1, 1024, 0, stream>>>(xWk, WrT, hseq_bf, out + 163840, out + 167936);
  // half1 = attended @ W1 + b1 ; half2 = x @ W1 + b1
  k_gemm_bias4<<<dim3(8, 32), blk, 0, stream>>>(att_bf,  W1T, b1, half1, 512, 512, 512, 512);
  k_gemm_bias4<<<dim3(8, 32), blk, 0, stream>>>(hseq_bf, W1T, b1, half2, 512, 512, 512, 512);
  // scores + softmax -> bf16 attention weights
  k_scores<<<256, 256, 0, stream>>>(half1, half2, W3, b3, attn_bf);
  // weighted = attn @ attended (batched)
  k_weighted4<<<dim3(8, 4, 8), blk, 0, stream>>>(attn_bf, attT, wgt_bf);
  // out = [x, weighted] @ Wout + bout
  k_outproj<<<dim3(1, 32), blk, 0, stream>>>(hseq_bf, wgt_bf, WoutT, bout, out);
}
